// Custom_Linear_48395691491501
// MI455X (gfx1250) — 8-bit matrix operand order repaired and checked by reading, compiled, not run on hardware
//
#include <hip/hip_runtime.h>
#include <stdint.h>

typedef int v8i __attribute__((ext_vector_type(8)));

#define M_DIM 8192
#define N_DIM 4096
#define K_DIM 4096

#define BM 128
#define BN 128
#define BK 64
#define ROWB 80

__device__ __forceinline__ void async_load_16B(uint32_t lds_off, const void* sbase,
                                               uint32_t goff) {
  asm volatile("global_load_async_to_lds_b128 %0, %1, %2"
               :: "v"(lds_off), "v"(goff), "s"((uint64_t)(uintptr_t)sbase)
               : "memory");
}

__device__ __forceinline__ void wait_async0() {
  asm volatile("s_wait_asynccnt 0" ::: "memory");
}

__global__ __launch_bounds__(256)
void int8_gemm_wmma_iu8(const int8_t* __restrict__ Aq,
                        const int8_t* __restrict__ Wq,
                        const float* __restrict__ p_sx,
                        const float* __restrict__ p_sw,
                        const float* __restrict__ p_so,
                        float* __restrict__ out) {
  __shared__ __align__(16) uint8_t ldsA[2][BM * ROWB];
  __shared__ __align__(16) uint8_t ldsB[2][BN * ROWB];

  const int tid  = threadIdx.x;
  const int lane = tid & 31;
  const int wid  = tid >> 5;
  const int l16  = lane & 15;
  const int lhi  = lane >> 4;

  const int bm0 = blockIdx.y * BM;
  const int bn0 = blockIdx.x * BN;

  const int waveM = wid & 3;
  const int waveN = wid >> 2;

  v8i acc[2][4];
  const v8i vzero = {};
  #pragma unroll
  for (int i = 0; i < 2; ++i)
    #pragma unroll
    for (int j = 0; j < 4; ++j) acc[i][j] = vzero;

  auto issue_tile = [&](int buf, int kq) {
    #pragma unroll
    for (int s = 0; s < 2; ++s) {
      const int idx = tid + s * 256;
      const int row = idx >> 2;
      const int kc  = (idx & 3) << 4;
      const uint32_t la = (uint32_t)(uintptr_t)&ldsA[buf][row * ROWB + kc];
      async_load_16B(la, Aq, (uint32_t)((bm0 + row) * K_DIM + kq + kc));
      const uint32_t lb = (uint32_t)(uintptr_t)&ldsB[buf][row * ROWB + kc];
      async_load_16B(lb, Wq, (uint32_t)((bn0 + row) * K_DIM + kq + kc));
    }
  };

  const int KT = K_DIM / BK;
  issue_tile(0, 0);

  for (int kt = 0; kt < KT; ++kt) {
    wait_async0();
    __syncthreads();
    if (kt + 1 < KT) issue_tile((kt + 1) & 1, (kt + 1) * BK);

    const uint8_t* Ab = ldsA[kt & 1];
    const uint8_t* Bb = ldsB[kt & 1];

    v8i bfrag[4];
    #pragma unroll
    for (int nt = 0; nt < 4; ++nt) {
      const uint8_t* bp = Bb + (waveN * 64 + nt * 16 + l16) * ROWB + lhi * 16;
      const int4 q0 = *(const int4*)(bp);
      const int4 q1 = *(const int4*)(bp + 32);
      v8i b;
      b[0] = q0.x; b[1] = q0.y; b[2] = q0.z; b[3] = q0.w;
      b[4] = q1.x; b[5] = q1.y; b[6] = q1.z; b[7] = q1.w;
      bfrag[nt] = b;
    }

    #pragma unroll
    for (int mt = 0; mt < 2; ++mt) {
      const uint8_t* ap = Ab + (waveM * 32 + mt * 16 + l16) * ROWB + lhi * 16;
      v8i a;
      #pragma unroll
      for (int j = 0; j < 4; ++j) {
        const int2 d = *(const int2*)(ap + 8 * (j & 1) + 32 * (j >> 1));
        a[2 * j] = d.x; a[2 * j + 1] = d.y;
      }
      #pragma unroll
      for (int nt = 0; nt < 4; ++nt) {
        acc[mt][nt] = __builtin_amdgcn_wmma_i32_16x16x64_iu8(
            true, a, true, bfrag[nt], acc[mt][nt], false, false);
      }
    }
  }

  const float scale = (p_sx[0] * p_sw[0]) / p_so[0];

  #pragma unroll
  for (int mt = 0; mt < 2; ++mt) {
    const int row0 = bm0 + waveM * 32 + mt * 16 + lhi * 8;
    #pragma unroll
    for (int nt = 0; nt < 4; ++nt) {
      const int col = bn0 + waveN * 64 + nt * 16 + l16;
      #pragma unroll
      for (int r = 0; r < 8; ++r) {
        float y = (float)acc[mt][nt][r] * scale;
        y = rintf(y);
        y = fminf(fmaxf(y, -128.0f), 127.0f);
        out[(size_t)(row0 + r) * N_DIM + col] = y;
      }
    }
  }

  if (blockIdx.x == 0 && blockIdx.y == 0 && tid == 0)
    out[(size_t)M_DIM * N_DIM] = p_so[0];
}

extern "C" void kernel_launch(void* const* d_in, const int* in_sizes, int n_in,
                              void* d_out, int out_size, void* d_ws, size_t ws_size,
                              hipStream_t stream) {
  (void)in_sizes; (void)n_in; (void)out_size; (void)d_ws; (void)ws_size;
  const int8_t* x  = (const int8_t*)d_in[0];
  const float*  sx = (const float*)d_in[1];
  const int8_t* w  = (const int8_t*)d_in[2];
  const float*  sw = (const float*)d_in[3];
  const float*  so = (const float*)d_in[4];
  float* out = (float*)d_out;

  dim3 grid(N_DIM / BN, M_DIM / BM);
  int8_gemm_wmma_iu8<<<grid, 256, 0, stream>>>(x, w, sx, sw, so, out);
}
